// ImplicitGeometricPriors_65326452572765
// MI455X (gfx1250) — matrix operand order repaired and checked by reading, not compiled, not run on hardware
//
#include <hip/hip_runtime.h>
#include <cmath>

typedef __attribute__((ext_vector_type(16))) _Float16 v16h;
typedef __attribute__((ext_vector_type(8)))  float    v8f;
typedef __attribute__((ext_vector_type(4)))  unsigned u32x4;
typedef __attribute__((ext_vector_type(8)))  int      i32x8;
typedef __attribute__((ext_vector_type(4)))  int      i32x4;

static constexpr int   BB   = 2;
static constexpr int   NP   = 2048;
static constexpr int   DD   = 256;
static constexpr int   HH   = 8;
static constexpr int   KK   = 64;
static constexpr int   HD   = 32;
static constexpr int   BN   = BB * NP;        // 4096
static constexpr long  BNK  = (long)BN * KK;  // 262144
static constexpr float SCL  = 0.17677669529663687f;  // 1/sqrt(32)

// ---------------------------------------------------------------------------
// 1. KNN: one block (256 threads) per query point. dist[2048] in LDS,
//    64 rounds of block-wide argmin with index tie-break (matches top_k order).
// ---------------------------------------------------------------------------
__global__ void knn_kernel(const float* __restrict__ xyz,
                           int* __restrict__ idxbuf, float* __restrict__ distbuf) {
  __shared__ float dist[NP];
  __shared__ float rval[256];
  __shared__ int   ridx[256];
  const int bn = blockIdx.x;            // 0..BN-1
  const int b  = bn / NP;
  const int t  = threadIdx.x;
  const float qx = xyz[bn * 3 + 0];
  const float qy = xyz[bn * 3 + 1];
  const float qz = xyz[bn * 3 + 2];
  const float* base = xyz + (long)b * NP * 3;
  for (int j = t; j < NP; j += 256) {
    float dx = qx - base[j * 3 + 0];
    float dy = qy - base[j * 3 + 1];
    float dz = qz - base[j * 3 + 2];
    float d2 = dx * dx + dy * dy + dz * dz;
    dist[j] = sqrtf(fmaxf(d2, 0.f));
  }
  __syncthreads();
  for (int s = 0; s < KK; ++s) {
    float bv = 3.0e38f; int bi = NP;
    for (int j = t; j < NP; j += 256) {
      float v = dist[j];
      if (v < bv || (v == bv && j < bi)) { bv = v; bi = j; }
    }
    rval[t] = bv; ridx[t] = bi;
    __syncthreads();
    for (int off = 128; off > 0; off >>= 1) {
      if (t < off) {
        float v2 = rval[t + off]; int i2 = ridx[t + off];
        if (v2 < rval[t] || (v2 == rval[t] && i2 < ridx[t])) { rval[t] = v2; ridx[t] = i2; }
      }
      __syncthreads();
    }
    if (t == 0) {
      int w = ridx[0];
      idxbuf[(long)bn * KK + s]  = w;
      distbuf[(long)bn * KK + s] = rval[0];
      dist[w] = 3.0e38f;
    }
    __syncthreads();
  }
}

// ---------------------------------------------------------------------------
// 2. Distance-encoding MLP + neighbor gather -> kv16 (f16 rows for WMMA GEMMs)
// ---------------------------------------------------------------------------
__global__ void de_mlp_kernel(const float* __restrict__ features,
                              const int* __restrict__ idxbuf,
                              const float* __restrict__ distbuf,
                              const float* __restrict__ w1, const float* __restrict__ b1,
                              const float* __restrict__ w2, const float* __restrict__ b2,
                              const float* __restrict__ w3, const float* __restrict__ b3,
                              _Float16* __restrict__ kv16) {
  __shared__ float h1s[8][64];
  __shared__ float h2s[8][128];
  const int t = threadIdx.x, w = t >> 5, lane = t & 31;
  const long r  = (long)blockIdx.x * 8 + w;   // row < BNK
  const long bn = r >> 6;
  const int  b  = (int)(bn / NP);
  const int  j  = idxbuf[r];
  const float d = distbuf[r];
  for (int u = 0; u < 2; ++u) {
    int e = lane + u * 32;
    h1s[w][e] = fmaxf(d * w1[e] + b1[e], 0.f);
  }
  __syncthreads();
  float h2v[4];
  for (int u = 0; u < 4; ++u) {
    int e = lane + u * 32;
    float acc = b2[e];
    for (int i = 0; i < 64; ++i) acc = fmaf(h1s[w][i], w2[i * 128 + e], acc);
    h2v[u] = fmaxf(acc, 0.f);
  }
  for (int u = 0; u < 4; ++u) h2s[w][lane + u * 32] = h2v[u];
  __syncthreads();
  const float* frow = features + ((long)b * NP + j) * DD;
  for (int u = 0; u < 8; ++u) {
    int e = lane + u * 32;
    float acc = b3[e];
    for (int i = 0; i < 128; ++i) acc = fmaf(h2s[w][i], w3[i * 256 + e], acc);
    kv16[r * DD + e] = (_Float16)(frow[e] + acc);
  }
}

// ---------------------------------------------------------------------------
// 3. Weight transpose f32[K,N] -> f16 BT[N,K] (k-contiguous for B fragments)
// ---------------------------------------------------------------------------
__global__ void transpose_w_kernel(const float* __restrict__ W, _Float16* __restrict__ BT,
                                   int Kd, int N) {
  long i = (long)blockIdx.x * 256 + threadIdx.x;
  if (i < (long)N * Kd) {
    int n = (int)(i / Kd), k = (int)(i % Kd);
    BT[i] = (_Float16)W[(long)k * N + n];
  }
}

__global__ void pack_feat_kernel(const float* __restrict__ features,
                                 _Float16* __restrict__ concat16) {
  long i = (long)blockIdx.x * 256 + threadIdx.x;   // over BN*DD
  long row = i >> 8; int c = (int)(i & 255);
  concat16[row * 512 + c] = (_Float16)features[i];
}

// ---------------------------------------------------------------------------
// TDM: issue one tensor_load_to_lds for a 2D tile of f16:
//   tileK halves (contiguous, dim0) x tileRows rows (dim1, stride = strideHalves)
// LDS destination rows are padded: 64B data + 16B pad (matches 40-half stride):
//   pad_interval code 3 -> every 16 DWORDs, pad_amount code 3 -> 4 DWORDs.
// D# layout per CDNA5 ISA (group0/group1).
// This toolchain exposes the 6-arg builtin:
//   (uint32x4 g0, int32x8 g1, int32x4 g2, int32x4 g3, int32x8 extra, i32 cpol)
// ---------------------------------------------------------------------------
__device__ __forceinline__ void tdm_load_tile_f16(unsigned ldsByteAddr,
                                                  const _Float16* gptr,
                                                  int tileK, int tileRows,
                                                  int strideHalves) {
  unsigned long long ga = (unsigned long long)(uintptr_t)gptr;
  u32x4 g0;
  g0[0] = 1u;                                   // count=1, user mode, no gather
  g0[1] = ldsByteAddr;                          // lds_addr (bytes)
  g0[2] = (unsigned)(ga & 0xFFFFFFFFull);       // global_addr[31:0]
  g0[3] = (unsigned)((ga >> 32) & 0x01FFFFFFull) | (2u << 30);  // addr[56:32] | type=2
  i32x8 g1;
  g1[0] = (int)((1u << 16)      // data_size = 1 -> 2 bytes
              | (1u << 20)      // pad_enable
              | (3u << 22)      // pad_interval: 16 DWORDs (64B) between pads
              | (3u << 25));    // pad_amount: 4 DWORDs (16B)
  g1[1] = (int)(((unsigned)tileK & 0xFFFFu) << 16);            // tensor_dim0[15:0]
  g1[2] = (int)((((unsigned)tileK >> 16) & 0xFFFFu)
              | (((unsigned)tileRows & 0xFFFFu) << 16));       // td0 hi | tensor_dim1 lo
  g1[3] = (int)((((unsigned)tileRows >> 16) & 0xFFFFu)
              | (((unsigned)tileK & 0xFFFFu) << 16));          // td1 hi | tile_dim0
  g1[4] = (int)((unsigned)tileRows & 0xFFFFu);                 // tile_dim1 | tile_dim2=0
  g1[5] = (int)(unsigned)strideHalves;                         // tensor_dim0_stride[31:0]
  g1[6] = 0;                                                   // stride hi | dim1_stride lo
  g1[7] = 0;
  i32x4 z4 = {0, 0, 0, 0};                                     // 2D tensor: groups 2/3 unused
  i32x8 z8 = {0, 0, 0, 0, 0, 0, 0, 0};
  __builtin_amdgcn_tensor_load_to_lds(g0, g1, z4, z4, z8, 0);
}

// ---------------------------------------------------------------------------
// 4. WMMA GEMM: C[M,N](f32|f16) = A[M,K]f16 @ BT[N,K]^T f16 + bias
//    128x128 block tile, 8 waves (2x4), K-step 32, v_wmma_f32_16x16x32_f16.
//    Tile staging via Tensor Data Mover (wave0 -> A tile, wave1 -> B tile).
// ---------------------------------------------------------------------------
union FragH { v16h v; unsigned u[8]; };

template <bool OUT_F16>
__global__ void gemm_wmma_kernel(const _Float16* __restrict__ A, int lda,
                                 const _Float16* __restrict__ BT,
                                 const float* __restrict__ bias,
                                 void* __restrict__ Cv, int ldc, int Kdim) {
  __shared__ alignas(16) _Float16 As[128 * 40];
  __shared__ alignas(16) _Float16 Bs[128 * 40];
  const int bm = blockIdx.x * 128, bn = blockIdx.y * 128;
  const int t = threadIdx.x;
  const int wave = t >> 5, lane = t & 31;
  const int wm = (wave >> 2) * 64;        // 0 | 64
  const int wn = (wave & 3) * 32;         // 0 | 32 | 64 | 96
  const int lrow = lane & 15, lhi = lane >> 4;
  const unsigned ldsA = (unsigned)(uintptr_t)&As[0];
  const unsigned ldsB = (unsigned)(uintptr_t)&Bs[0];

  v8f acc[4][2] = {};

  for (int k0 = 0; k0 < Kdim; k0 += 32) {
    if (k0 + 32 < Kdim) {   // prefetch next K tile into GL2 (global_prefetch_b8)
      __builtin_prefetch(&A[(long)(bm + (t >> 1)) * lda + k0 + 32], 0, 1);
      __builtin_prefetch(&BT[(long)(bn + (t >> 1)) * Kdim + k0 + 32], 0, 1);
    }
    // Tensor Data Mover stages both tiles; one DMA op per tile.
    if (wave == 0)
      tdm_load_tile_f16(ldsA, &A[(long)bm * lda + k0], 32, 128, lda);
    else if (wave == 1)
      tdm_load_tile_f16(ldsB, &BT[(long)bn * Kdim + k0], 32, 128, Kdim);
    if (wave < 2)
      __builtin_amdgcn_s_wait_tensorcnt(0);
    __syncthreads();

    // A fragments: lane l rows m, VGPR j -> K = (j>>2)*16 + (l>=16?8:0) + (j&3)*2
    FragH af[4];
#pragma unroll
    for (int mt = 0; mt < 4; ++mt) {
      int row = wm + mt * 16 + lrow;
#pragma unroll
      for (int j = 0; j < 8; ++j) {
        int kk = ((j >> 2) << 4) + (lhi << 3) + ((j & 3) << 1);
        af[mt].u[j] = *(const unsigned*)&As[row * 40 + kk];
      }
    }
    // B fragments: lane l holds column (l&15), VGPR j -> K = (j>>2)*16 + (l>=16?8:0) + (j&3)*2
    FragH bf[2];
#pragma unroll
    for (int nt = 0; nt < 2; ++nt) {
      int col = wn + nt * 16 + lrow;
      int kh = lhi << 4;
#pragma unroll
      for (int j = 0; j < 8; ++j)
        bf[nt].u[j] = *(const unsigned*)&Bs[col * 40 + ((j >> 2) << 4) + (lhi << 3) + ((j & 3) << 1)];
    }
#pragma unroll
    for (int mt = 0; mt < 4; ++mt)
#pragma unroll
      for (int nt = 0; nt < 2; ++nt)
        acc[mt][nt] = __builtin_amdgcn_wmma_f32_16x16x32_f16(
            false, af[mt].v, false, bf[nt].v, (short)0, acc[mt][nt], false, false);
    __syncthreads();
  }

  // Epilogue: C/D layout lane(l&15)=col, VGPR r -> row r (+8 for lanes 16-31)
#pragma unroll
  for (int mt = 0; mt < 4; ++mt)
#pragma unroll
    for (int nt = 0; nt < 2; ++nt)
#pragma unroll
      for (int r = 0; r < 8; ++r) {
        int row = bm + wm + mt * 16 + (lhi << 3) + r;
        int col = bn + wn + nt * 16 + lrow;
        float vv = acc[mt][nt][r] + bias[col];
        if constexpr (OUT_F16)
          ((_Float16*)Cv)[(long)row * ldc + col] = (_Float16)vv;
        else
          ((float*)Cv)[(long)row * ldc + col] = vv;
      }
}

// ---------------------------------------------------------------------------
// 5. Per-point attention: one wave per (b,n,h). 64 keys, shfl softmax.
// ---------------------------------------------------------------------------
__global__ void attn_kernel(const float* __restrict__ qbuf,
                            const _Float16* __restrict__ kbuf,
                            const _Float16* __restrict__ vbuf,
                            _Float16* __restrict__ ctx16) {
  __shared__ float attn_lds[8 * 64];
  const int t = threadIdx.x, w = t >> 5, lane = t & 31;
  const int gid = blockIdx.x * 8 + w;     // < BN*HH
  const int bn = gid >> 3, h = gid & 7;
  const float* q = qbuf + (long)bn * DD + h * HD;
  const _Float16* kb = kbuf + (long)bn * KK * DD + h * HD;
  float s0 = 0.f, s1 = 0.f;
  for (int d = 0; d < HD; ++d) {
    float qd = q[d];
    s0 = fmaf(qd, (float)kb[(long)lane * DD + d], s0);
    s1 = fmaf(qd, (float)kb[(long)(lane + 32) * DD + d], s1);
  }
  s0 *= SCL; s1 *= SCL;
  float m = fmaxf(s0, s1);
  for (int off = 16; off > 0; off >>= 1) m = fmaxf(m, __shfl_xor(m, off));
  float e0 = __expf(s0 - m), e1 = __expf(s1 - m);
  float sum = e0 + e1;
  for (int off = 16; off > 0; off >>= 1) sum += __shfl_xor(sum, off);
  float inv = 1.f / sum;
  attn_lds[w * 64 + lane]      = e0 * inv;
  attn_lds[w * 64 + lane + 32] = e1 * inv;
  __syncthreads();
  const _Float16* vb = vbuf + (long)bn * KK * DD + h * HD;
  float acc = 0.f;
  for (int kk = 0; kk < KK; ++kk)
    acc = fmaf(attn_lds[w * 64 + kk], (float)vb[(long)kk * DD + lane], acc);
  ctx16[(long)bn * DD + h * HD + lane] = (_Float16)acc;
}

// ---------------------------------------------------------------------------
// 6. LayerNorm + ReLU -> f16 (block per row, 256 threads)
// ---------------------------------------------------------------------------
__global__ void ln_relu_kernel(const float* __restrict__ x,
                               const float* __restrict__ g,
                               const float* __restrict__ bta,
                               _Float16* __restrict__ out) {
  __shared__ float red[256];
  const int row = blockIdx.x, t = threadIdx.x;
  float v = x[(long)row * DD + t];
  red[t] = v; __syncthreads();
  for (int off = 128; off > 0; off >>= 1) { if (t < off) red[t] += red[t + off]; __syncthreads(); }
  float mu = red[0] * (1.f / 256.f);
  __syncthreads();
  float dv = v - mu;
  red[t] = dv * dv; __syncthreads();
  for (int off = 128; off > 0; off >>= 1) { if (t < off) red[t] += red[t + off]; __syncthreads(); }
  float var = red[0] * (1.f / 256.f);
  float y = dv * rsqrtf(var + 1e-5f) * g[t] + bta[t];
  out[(long)row * DD + t] = (_Float16)fmaxf(y, 0.f);
}

// ---------------------------------------------------------------------------
extern "C" void kernel_launch(void* const* d_in, const int* in_sizes, int n_in,
                              void* d_out, int out_size, void* d_ws, size_t ws_size,
                              hipStream_t stream) {
  const float* features = (const float*)d_in[0];
  const float* xyz      = (const float*)d_in[1];
  const float* de_w1    = (const float*)d_in[2];
  const float* de_b1    = (const float*)d_in[3];
  const float* de_w2    = (const float*)d_in[4];
  const float* de_b2    = (const float*)d_in[5];
  const float* de_w3    = (const float*)d_in[6];
  const float* de_b3    = (const float*)d_in[7];
  const float* wq = (const float*)d_in[8];  const float* bq = (const float*)d_in[9];
  const float* wk = (const float*)d_in[10]; const float* bk = (const float*)d_in[11];
  const float* wv = (const float*)d_in[12]; const float* bv = (const float*)d_in[13];
  const float* wo = (const float*)d_in[14]; const float* bo = (const float*)d_in[15];
  const float* se_w1 = (const float*)d_in[16]; const float* se_b1 = (const float*)d_in[17];
  const float* ln_g  = (const float*)d_in[18]; const float* ln_b  = (const float*)d_in[19];
  const float* se_w2 = (const float*)d_in[20]; const float* se_b2 = (const float*)d_in[21];
  (void)in_sizes; (void)n_in; (void)out_size; (void)ws_size;

  char* ws = (char*)d_ws;
  size_t off = 0;
  auto carve = [&](size_t bytes) -> void* {
    void* p = ws + off;
    off = (off + bytes + 255) & ~(size_t)255;
    return p;
  };
  int*      idxbuf  = (int*)     carve(BNK * sizeof(int));
  float*    distbuf = (float*)   carve(BNK * sizeof(float));
  _Float16* kv16    = (_Float16*)carve(BNK * DD * sizeof(_Float16));   // 128 MB
  _Float16* kbuf    = (_Float16*)carve(BNK * DD * sizeof(_Float16));   // 128 MB
  _Float16* vbuf    = (_Float16*)carve(BNK * DD * sizeof(_Float16));   // 128 MB
  float*    qbuf    = (float*)   carve((long)BN * DD * sizeof(float));
  _Float16* ctx16   = (_Float16*)carve((long)BN * DD * sizeof(_Float16));
  _Float16* concat16= (_Float16*)carve((long)BN * 2 * DD * sizeof(_Float16));
  float*    xbuf    = (float*)   carve((long)BN * DD * sizeof(float));
  _Float16* ln16    = (_Float16*)carve((long)BN * DD * sizeof(_Float16));
  _Float16* wqt  = (_Float16*)carve(DD * DD * sizeof(_Float16));
  _Float16* wkt  = (_Float16*)carve(DD * DD * sizeof(_Float16));
  _Float16* wvt  = (_Float16*)carve(DD * DD * sizeof(_Float16));
  _Float16* wot  = (_Float16*)carve(DD * DD * sizeof(_Float16));
  _Float16* se1t = (_Float16*)carve(2 * DD * DD * sizeof(_Float16));
  _Float16* se2t = (_Float16*)carve(DD * DD * sizeof(_Float16));

  // 1. KNN
  knn_kernel<<<BN, 256, 0, stream>>>(xyz, idxbuf, distbuf);

  // 2. features -> concat16 cols [0,256)
  pack_feat_kernel<<<(BN * DD) / 256, 256, 0, stream>>>(features, concat16);

  // 3. weight transposes (f32 [K,N] -> f16 [N,K])
  auto T = [&](const float* W, _Float16* BT, int Kd, int N) {
    transpose_w_kernel<<<(Kd * N + 255) / 256, 256, 0, stream>>>(W, BT, Kd, N);
  };
  T(wq, wqt, DD, DD); T(wk, wkt, DD, DD); T(wv, wvt, DD, DD); T(wo, wot, DD, DD);
  T(se_w1, se1t, 2 * DD, DD); T(se_w2, se2t, DD, DD);

  // 4. distance MLP + gather -> kv16
  de_mlp_kernel<<<(int)(BNK / 8), 256, 0, stream>>>(
      features, idxbuf, distbuf, de_w1, de_b1, de_w2, de_b2, de_w3, de_b3, kv16);

  auto G = [&](const _Float16* A, int lda, const _Float16* BT, const float* bias,
               void* C, int ldc, int M, int N, int Kd, bool outf16) {
    dim3 grid(M / 128, N / 128);
    if (outf16) gemm_wmma_kernel<true><<<grid, 256, 0, stream>>>(A, lda, BT, bias, C, ldc, Kd);
    else        gemm_wmma_kernel<false><<<grid, 256, 0, stream>>>(A, lda, BT, bias, C, ldc, Kd);
  };

  // 5. q = features @ wq + bq  (A = concat16 cols 0..255, lda=512)
  G(concat16, 2 * DD, wqt, bq, qbuf, DD, BN, DD, DD, false);
  // 6. k = kv @ wk + bk ; v = kv @ wv + bv  (f16 outputs)
  G(kv16, DD, wkt, bk, kbuf, DD, (int)BNK, DD, DD, true);
  G(kv16, DD, wvt, bv, vbuf, DD, (int)BNK, DD, DD, true);
  // 7. attention
  attn_kernel<<<(BN * HH) / 8, 256, 0, stream>>>(qbuf, kbuf, vbuf, ctx16);
  // 8. attended = ctx @ wo + bo  -> concat16 cols [256,512)
  G(ctx16, DD, wot, bo, (void*)(concat16 + DD), 2 * DD, BN, DD, DD, true);
  // 9. x = concat @ se_w1 + se_b1
  G(concat16, 2 * DD, se1t, se_b1, xbuf, DD, BN, DD, 2 * DD, false);
  // 10. LayerNorm + ReLU
  ln_relu_kernel<<<BN, 256, 0, stream>>>(xbuf, ln_g, ln_b, ln16);
  // 11. out = relu(ln(x)) @ se_w2 + se_b2
  G(ln16, DD, se2t, se_b2, d_out, DD, BN, DD, DD, false);
}
